// TransformerBlock_11441792876641
// MI455X (gfx1250) — matrix operand order repaired and checked by reading, not compiled, not run on hardware
//
#include <hip/hip_runtime.h>
#include <hip/hip_bf16.h>
#include <math.h>

// ---------------- types ----------------
typedef __attribute__((ext_vector_type(16))) __bf16 v16bf;
typedef __attribute__((ext_vector_type(8)))  __bf16 v8bf;
typedef __attribute__((ext_vector_type(8)))  float  v8f;

#define NTOK 32768   // B*T = 8192*4 tokens

__device__ __forceinline__ float sigmoidf_(float x) { return 1.0f / (1.0f + __expf(-x)); }
__device__ __forceinline__ float gelu_exact(float x) {
    return 0.5f * x * (1.0f + erff(x * 0.70710678118654752f));
}

// epilogue applicator (32-bit offsets: max index 32768*1024 < 2^31)
template <int MODE>
__device__ __forceinline__ void epi(float* __restrict__ out, const float* __restrict__ aux,
                                    const float* __restrict__ res, unsigned o, float v)
{
    if      (MODE == 0) out[o] = v;
    else if (MODE == 1) out[o] += v;
    else if (MODE == 2) out[o] += v * aux[o];
    else if (MODE == 3) out[o] = gelu_exact(v);
    else                out[o] = res[o] + v;
}

// ---------------- WMMA GEMM: out[N,M] (+)= A[N,K] @ W[K,M] + bias, fused epilogues ---
// MODE 0: out = acc+b ; 1: out += acc+b ; 2: out += (acc+b)*aux ; 3: out = gelu(acc+b) ;
// MODE 4: out = res + acc+b
// Block: 256 thr = 8 waves; block tile 128x64; wave tile 32x32 (4 wmma/K-step);
// K step 32; LDS double-buffered (one barrier per K-step).
template <int MODE>
__global__ __launch_bounds__(256)
void gemm_bf16_wmma(const float* __restrict__ A, const float* __restrict__ W,
                    const float* __restrict__ bias, float* __restrict__ out,
                    const float* __restrict__ aux, const float* __restrict__ res,
                    int K, int M)
{
    __shared__ __bf16 lA[2][128][48];   // ping-pong, 16B-aligned rows (24KB)
    __shared__ __bf16 lB[2][64][48];    // stored transposed: lB[.][n][k] (12KB)

    const int t    = threadIdx.x;
    const int lane = t & 31;
    const int wave = t >> 5;
    const unsigned rowBlk = blockIdx.y * 128;
    const unsigned colBlk = blockIdx.x * 64;
    const int wrow = (wave >> 1) * 32;   // 0,32,64,96
    const int wcol = (wave & 1) * 32;    // 0,32

    // global A loader: row ar (0..127), 16 floats starting at ac
    const int ar = t >> 1;
    const int ac = (t & 1) * 16;
    // global B loader: k row bkk (0..31), 8 floats starting at bm8
    const int bkk = t >> 3;
    const int bm8 = (t & 7) * 8;

    const int khalf = lane >> 4;   // operand K-half select
    const int mrow  = lane & 15;   // A row within tile / B,C col within tile

    const unsigned Mu = (unsigned)M;
    const unsigned aRowOff = (rowBlk + ar) * (unsigned)K + ac;  // + k0 each tile
    const unsigned bOff    = bkk * Mu + colBlk + bm8;           // + k0*M each tile

    v8f acc00 = {}, acc01 = {}, acc10 = {}, acc11 = {};

    float4 ra0, ra1, ra2, ra3, rb0, rb1;

    auto loadG = [&](int k0) {
        const float4* ga = (const float4*)(A + aRowOff + (unsigned)k0);
        ra0 = ga[0]; ra1 = ga[1]; ra2 = ga[2]; ra3 = ga[3];
        const float4* gb = (const float4*)(W + bOff + (unsigned)k0 * Mu);
        rb0 = gb[0]; rb1 = gb[1];
    };
    auto storeL = [&](int buf) {
        float af[16] = { ra0.x, ra0.y, ra0.z, ra0.w, ra1.x, ra1.y, ra1.z, ra1.w,
                         ra2.x, ra2.y, ra2.z, ra2.w, ra3.x, ra3.y, ra3.z, ra3.w };
        #pragma unroll
        for (int j = 0; j < 16; ++j) lA[buf][ar][ac + j] = (__bf16)af[j];
        float bf[8] = { rb0.x, rb0.y, rb0.z, rb0.w, rb1.x, rb1.y, rb1.z, rb1.w };
        #pragma unroll
        for (int j = 0; j < 8; ++j) lB[buf][bm8 + j][bkk] = (__bf16)bf[j];
    };
    auto compute = [&](int buf) {
        // CDNA5 16-bit operand layouts (ISA 7.12.2):
        // A 16x32: lanes<16 hold K 0-7 & 16-23, lanes>=16 hold K 8-15 & 24-31
        v8bf a0lo = *(const v8bf*)&lA[buf][wrow + mrow][khalf * 8];
        v8bf a0hi = *(const v8bf*)&lA[buf][wrow + mrow][16 + khalf * 8];
        v16bf fragA0 = __builtin_shufflevector(a0lo, a0hi, 0,1,2,3,4,5,6,7,8,9,10,11,12,13,14,15);
        v8bf a1lo = *(const v8bf*)&lA[buf][wrow + 16 + mrow][khalf * 8];
        v8bf a1hi = *(const v8bf*)&lA[buf][wrow + 16 + mrow][16 + khalf * 8];
        v16bf fragA1 = __builtin_shufflevector(a1lo, a1hi, 0,1,2,3,4,5,6,7,8,9,10,11,12,13,14,15);
        // B 32x16: lanes<16 hold K 0-7 & 16-23, lanes>=16 hold K 8-15 & 24-31 (col = lane&15)
        v8bf b0lo = *(const v8bf*)&lB[buf][wcol + mrow][khalf * 8];
        v8bf b0hi = *(const v8bf*)&lB[buf][wcol + mrow][16 + khalf * 8];
        v16bf fragB0 = __builtin_shufflevector(b0lo, b0hi, 0,1,2,3,4,5,6,7,8,9,10,11,12,13,14,15);
        v8bf b1lo = *(const v8bf*)&lB[buf][wcol + 16 + mrow][khalf * 8];
        v8bf b1hi = *(const v8bf*)&lB[buf][wcol + 16 + mrow][16 + khalf * 8];
        v16bf fragB1 = __builtin_shufflevector(b1lo, b1hi, 0,1,2,3,4,5,6,7,8,9,10,11,12,13,14,15);

        acc00 = __builtin_amdgcn_wmma_f32_16x16x32_bf16(false, fragA0, false, fragB0,
                                                        (short)0, acc00, false, false);
        acc01 = __builtin_amdgcn_wmma_f32_16x16x32_bf16(false, fragA0, false, fragB1,
                                                        (short)0, acc01, false, false);
        acc10 = __builtin_amdgcn_wmma_f32_16x16x32_bf16(false, fragA1, false, fragB0,
                                                        (short)0, acc10, false, false);
        acc11 = __builtin_amdgcn_wmma_f32_16x16x32_bf16(false, fragA1, false, fragB1,
                                                        (short)0, acc11, false, false);
    };

    // ---- software pipeline: one barrier per K-step ----
    loadG(0);
    storeL(0);
    __syncthreads();
    const int nk = K >> 5;
    for (int i = 0; i < nk - 1; ++i) {
        const int k0 = (i + 1) << 5;
        loadG(k0);                                   // next tile -> regs
        const int kp = (k0 + 32 < K) ? (k0 + 32) : k0;  // branchless lookahead
        __builtin_prefetch(A + aRowOff + (unsigned)kp, 0, 1);
        __builtin_prefetch(W + bOff + (unsigned)kp * Mu, 0, 1);
        compute(i & 1);                              // current tile
        storeL((i + 1) & 1);                         // stage next tile
        __syncthreads();
    }
    compute((nk - 1) & 1);

    // ---- epilogue: D layout -> row = r + 8*(lane>>4), col = lane&15 (all 32-bit) ----
    const unsigned col0 = colBlk + wcol + mrow;
    const unsigned col1 = col0 + 16;
    const float bi0 = bias[col0];
    const float bi1 = bias[col1];
    #pragma unroll
    for (int r = 0; r < 8; ++r) {
        const unsigned row0 = rowBlk + wrow + r + 8 * khalf;
        const unsigned row1 = row0 + 16;
        const unsigned base0 = row0 * Mu;
        const unsigned base1 = row1 * Mu;
        epi<MODE>(out, aux, res, base0 + col0, acc00[r] + bi0);
        epi<MODE>(out, aux, res, base0 + col1, acc01[r] + bi1);
        epi<MODE>(out, aux, res, base1 + col0, acc10[r] + bi0);
        epi<MODE>(out, aux, res, base1 + col1, acc11[r] + bi1);
    }
}

// ---------------- wave-per-row LayerNorm over last dim D ----------------
__global__ __launch_bounds__(256)
void ln_rows(const float* __restrict__ in, const float* __restrict__ g,
             const float* __restrict__ b, float* __restrict__ out, int D)
{
    const int lane = threadIdx.x & 31;
    const unsigned row = blockIdx.x * 8 + (threadIdx.x >> 5);
    const float* x = in + (size_t)row * D;
    float* o = out + (size_t)row * D;
    float s = 0.f, ss = 0.f;
    for (int i = lane; i < D; i += 32) { float v = x[i]; s += v; ss += v * v; }
    #pragma unroll
    for (int off = 16; off > 0; off >>= 1) {
        s  += __shfl_xor(s,  off, 32);
        ss += __shfl_xor(ss, off, 32);
    }
    const float mean = s / (float)D;
    const float var  = ss / (float)D - mean * mean;
    const float inv  = rsqrtf(var + 1e-5f);
    for (int i = lane; i < D; i += 32) o[i] = (x[i] - mean) * inv * g[i] + b[i];
}

// ---------------- per-batch attention, T=4, DK=256; wave per query token ----------------
__global__ __launch_bounds__(128)
void attn_t4(const float* __restrict__ q, const float* __restrict__ k,
             const float* __restrict__ v, float* __restrict__ fused,
             int fusedOff, float* __restrict__ Aout, int branch)
{
    const int b    = blockIdx.x;
    const int i    = threadIdx.x >> 5;     // query token 0..3
    const int lane = threadIdx.x & 31;
    const unsigned base = (unsigned)b * 4u * 256u;

    float qi[8];
    #pragma unroll
    for (int e = 0; e < 8; ++e) qi[e] = q[base + i * 256u + lane + e * 32];

    float sc[4];
    #pragma unroll
    for (int j = 0; j < 4; ++j) {
        float p = 0.f;
        #pragma unroll
        for (int e = 0; e < 8; ++e) p += qi[e] * k[base + j * 256u + lane + e * 32];
        #pragma unroll
        for (int off = 16; off > 0; off >>= 1) p += __shfl_xor(p, off, 32);
        sc[j] = p * 0.0625f;               // 1/sqrt(256)
    }
    float mx = fmaxf(fmaxf(sc[0], sc[1]), fmaxf(sc[2], sc[3]));
    float a[4]; float sum = 0.f;
    #pragma unroll
    for (int j = 0; j < 4; ++j) { a[j] = __expf(sc[j] - mx); sum += a[j]; }
    const float rs = 1.0f / sum;
    #pragma unroll
    for (int j = 0; j < 4; ++j) a[j] *= rs;

    #pragma unroll
    for (int e = 0; e < 8; ++e) {
        float o = 0.f;
        #pragma unroll
        for (int j = 0; j < 4; ++j) o += a[j] * v[base + j * 256u + lane + e * 32];
        fused[(unsigned)(b * 4 + i) * 512u + fusedOff + lane + e * 32] = o;
    }
    if (lane < 4)  // A[b, branch, i, 0, lane]
        Aout[((unsigned)b * 2 + branch) * 16u + i * 4 + lane] = a[lane];
}

// ---------------- mLSTM gate combine: I <- sigmoid(F)*C + sigmoid(I)*Z ----------------
__global__ __launch_bounds__(256)
void lstm_combine(float* __restrict__ I, const float* __restrict__ F,
                  const float* __restrict__ Z, const float* __restrict__ C)
{
    const unsigned idx = blockIdx.x * 256u + threadIdx.x;
    I[idx] = sigmoidf_(F[idx]) * C[idx] + sigmoidf_(I[idx]) * Z[idx];
}

// ---------------- host-side orchestration ----------------
static void gemm(int mode, const float* A, const float* W, const float* bias,
                 float* out, const float* aux, const float* res,
                 int K, int M, hipStream_t s)
{
    dim3 g(M / 64, NTOK / 128), blk(256);
    switch (mode) {
    case 0: gemm_bf16_wmma<0><<<g, blk, 0, s>>>(A, W, bias, out, aux, res, K, M); break;
    case 1: gemm_bf16_wmma<1><<<g, blk, 0, s>>>(A, W, bias, out, aux, res, K, M); break;
    case 2: gemm_bf16_wmma<2><<<g, blk, 0, s>>>(A, W, bias, out, aux, res, K, M); break;
    case 3: gemm_bf16_wmma<3><<<g, blk, 0, s>>>(A, W, bias, out, aux, res, K, M); break;
    case 4: gemm_bf16_wmma<4><<<g, blk, 0, s>>>(A, W, bias, out, aux, res, K, M); break;
    }
}

static void ln(const float* in, const float* g, const float* b, float* out, int D, hipStream_t s)
{
    ln_rows<<<NTOK / 8, 256, 0, s>>>(in, g, b, out, D);
}

extern "C" void kernel_launch(void* const* d_in, const int* in_sizes, int n_in,
                              void* d_out, int out_size, void* d_ws, size_t ws_size,
                              hipStream_t stream)
{
    auto IN = [&](int i) { return (const float*)d_in[i]; };
    // ---- input leaf indices (setup_inputs dict insertion order, depth-first) ----
    const float* state_i = IN(0);    // [N,256]
    const float* C_mu    = IN(1);    // [N,1024]
    const float* C_q     = IN(2);    // [N,1024]
    // 3,4 norm_attn g/b ; 5..10 Wq/Wk/Wv w,b ; 11..16 ln_q/k/v g,b
    // 17..20 ln_film_mu g,b ln_film_q g,b
    // 21..32 film_mu {qg w,b; qb w,b; kg; kb; vg; vb} ; 33..44 film_q
    // 45,46 fuse ; 47,48 norm_ffn ; 49..52 ffn_w1,ffn_w2
    // 53..68 lstm_mu {WI,WF,WZ,RI,RF,RZ (w,b)x6, norm_c g,b, c_proj w,b} ; 69..84 lstm_q

    // ---- workspace carve (fp32 throughout) ----
    char* wp = (char*)d_ws;
    auto carve = [&](size_t elems) { float* p = (float*)wp; wp += ((elems * 4 + 255) & ~(size_t)255); return p; };
    const size_t N = NTOK;
    float* x    = carve(N * 256);
    float* q    = carve(N * 256);
    float* k    = carve(N * 256);
    float* v    = carve(N * 256);
    float* hmu  = carve(N * 1024);
    float* hq   = carve(N * 1024);
    float* qm   = carve(N * 256);
    float* km   = carve(N * 256);
    float* vm   = carve(N * 256);
    float* qq   = carve(N * 256);
    float* kq   = carve(N * 256);
    float* vq   = carve(N * 256);
    float* fused= carve(N * 512);
    float* h    = carve(N * 256);
    float* xff  = carve(N * 256);
    float* mid  = carve(N * 1024);
    float* h2   = carve(N * 256);
    float* gI   = carve(N * 1024);
    float* gF   = carve(N * 1024);
    float* gZ   = carve(N * 1024);
    float* cln  = carve(N * 1024);

    float* outC_mu = (float*)d_out;
    float* outC_q  = outC_mu + N * 1024;
    float* outA    = outC_q + N * 1024;

    // ---- attention input path ----
    ln(state_i, IN(3), IN(4), x, 256, stream);                       // x = LN(state_i)
    gemm(0, x, IN(5), IN(6), q, nullptr, nullptr, 256, 256, stream); // q = x@Wq+b
    gemm(0, x, IN(7), IN(8), k, nullptr, nullptr, 256, 256, stream);
    gemm(0, x, IN(9), IN(10), v, nullptr, nullptr, 256, 256, stream);
    ln(q, IN(11), IN(12), q, 256, stream);                           // per-head LN (in place)
    ln(k, IN(13), IN(14), k, 256, stream);
    ln(v, IN(15), IN(16), v, 256, stream);
    ln(C_mu, IN(17), IN(18), hmu, 1024, stream);                     // FiLM conditioners
    ln(C_q,  IN(19), IN(20), hq,  1024, stream);

    // ---- FiLM: out = q + (h@Wg+bg)*q + (h@Wb+bb), fused in GEMM epilogues ----
    const size_t bytes256 = (size_t)N * 256 * sizeof(float);
    hipMemcpyAsync(qm, q, bytes256, hipMemcpyDeviceToDevice, stream);
    hipMemcpyAsync(km, k, bytes256, hipMemcpyDeviceToDevice, stream);
    hipMemcpyAsync(vm, v, bytes256, hipMemcpyDeviceToDevice, stream);
    hipMemcpyAsync(qq, q, bytes256, hipMemcpyDeviceToDevice, stream);
    hipMemcpyAsync(kq, k, bytes256, hipMemcpyDeviceToDevice, stream);
    hipMemcpyAsync(vq, v, bytes256, hipMemcpyDeviceToDevice, stream);
    gemm(2, hmu, IN(21), IN(22), qm, q, nullptr, 1024, 256, stream); // += gamma*q
    gemm(1, hmu, IN(23), IN(24), qm, nullptr, nullptr, 1024, 256, stream); // += beta
    gemm(2, hmu, IN(25), IN(26), km, k, nullptr, 1024, 256, stream);
    gemm(1, hmu, IN(27), IN(28), km, nullptr, nullptr, 1024, 256, stream);
    gemm(2, hmu, IN(29), IN(30), vm, v, nullptr, 1024, 256, stream);
    gemm(1, hmu, IN(31), IN(32), vm, nullptr, nullptr, 1024, 256, stream);
    gemm(2, hq,  IN(33), IN(34), qq, q, nullptr, 1024, 256, stream);
    gemm(1, hq,  IN(35), IN(36), qq, nullptr, nullptr, 1024, 256, stream);
    gemm(2, hq,  IN(37), IN(38), kq, k, nullptr, 1024, 256, stream);
    gemm(1, hq,  IN(39), IN(40), kq, nullptr, nullptr, 1024, 256, stream);
    gemm(2, hq,  IN(41), IN(42), vq, v, nullptr, 1024, 256, stream);
    gemm(1, hq,  IN(43), IN(44), vq, nullptr, nullptr, 1024, 256, stream);

    // ---- two attentions (writes concat buffer + A output) ----
    attn_t4<<<8192, 128, 0, stream>>>(qm, km, vm, fused, 0,   outA, 0);
    attn_t4<<<8192, 128, 0, stream>>>(qq, kq, vq, fused, 256, outA, 1);

    // ---- fuse + residual, FFN ----
    gemm(4, fused, IN(45), IN(46), h, nullptr, state_i, 512, 256, stream);  // h = state_i + fuse(...)
    ln(h, IN(47), IN(48), xff, 256, stream);
    gemm(3, xff, IN(49), IN(50), mid, nullptr, nullptr, 256, 1024, stream); // gelu(x@w1+b1)
    gemm(4, mid, IN(51), IN(52), h2, nullptr, h, 1024, 256, stream);        // h2 = h + mid@w2+b2

    // ---- mLSTM cells ----
    const int numBlk1024 = (int)((N * 1024) / 256);
    for (int br = 0; br < 2; ++br) {
        const int base = 53 + br * 16;
        const float* C = (br == 0) ? C_mu : C_q;
        float* outC = (br == 0) ? outC_mu : outC_q;
        // gates: G = C@W* + h2@R*  (two accumulating GEMMs each)
        gemm(0, C,  IN(base + 0), IN(base + 1), gI, nullptr, nullptr, 1024, 1024, stream); // WI
        gemm(1, h2, IN(base + 6), IN(base + 7), gI, nullptr, nullptr, 256, 1024, stream);  // RI
        gemm(0, C,  IN(base + 2), IN(base + 3), gF, nullptr, nullptr, 1024, 1024, stream); // WF
        gemm(1, h2, IN(base + 8), IN(base + 9), gF, nullptr, nullptr, 256, 1024, stream);  // RF
        gemm(0, C,  IN(base + 4), IN(base + 5), gZ, nullptr, nullptr, 1024, 1024, stream); // WZ
        gemm(1, h2, IN(base +10), IN(base +11), gZ, nullptr, nullptr, 256, 1024, stream);  // RZ
        // c_pre = sig(F)*C + sig(I)*Z  (into gI)
        lstm_combine<<<numBlk1024, 256, 0, stream>>>(gI, gF, gZ, C);
        ln(gI, IN(base + 12), IN(base + 13), cln, 1024, stream);            // norm_c
        gemm(3, cln, IN(base + 14), IN(base + 15), outC, nullptr, nullptr, 1024, 1024, stream); // gelu(c_proj)
    }

    (void)in_sizes; (void)n_in; (void)out_size; (void)ws_size;
}
